// MultiHeadAttention_30030411334002
// MI455X (gfx1250) — matrix operand order repaired and checked by reading, not compiled, not run on hardware
//
#include <hip/hip_runtime.h>
#include <math.h>

// ---------------------------------------------------------------------------
// MI455X / gfx1250 multi-head attention, wave32 + WMMA f32_16x16x32_f16.
// B=2, S=2048, D=1024, H=16, Dh=64.
// ---------------------------------------------------------------------------

typedef __attribute__((ext_vector_type(16))) _Float16 v16h;
typedef __attribute__((ext_vector_type(8)))  float    v8f;

union Frag {
  v16h v;
  _Float16 h[16];
  unsigned int u[8];
  uint4 q[2];
};

__device__ __forceinline__ unsigned int pack2h(float a, float b) {
  union { _Float16 h[2]; unsigned int u; } t;
  t.h[0] = (_Float16)a;
  t.h[1] = (_Float16)b;
  return t.u;
}

// ---------------------------------------------------------------------------
// fp32 -> f16 conversion
// ---------------------------------------------------------------------------
__global__ void cvt_f16_kernel(const float* __restrict__ in,
                               _Float16* __restrict__ out, int n) {
  int i = blockIdx.x * blockDim.x + threadIdx.x;
  if (i < n) out[i] = (_Float16)in[i];
}

// in: [K][N] fp32 row-major -> out: [N][K] f16 row-major. Tiled via LDS so
// both global sides stay coalesced. Grid (N/32, K/32), 256 threads.
__global__ __launch_bounds__(256) void transpose_f16_kernel(
    const float* __restrict__ in, _Float16* __restrict__ out, int K, int N) {
  __shared__ _Float16 tile[32][33];  // +1 pad: no LDS bank conflicts
  const int bx = blockIdx.x * 32;    // N offset
  const int by = blockIdx.y * 32;    // K offset
  const int tx = threadIdx.x & 31;
  const int ty = threadIdx.x >> 5;   // 0..7
#pragma unroll
  for (int j = 0; j < 32; j += 8)
    tile[ty + j][tx] = (_Float16)in[(size_t)(by + ty + j) * N + bx + tx];
  __syncthreads();
#pragma unroll
  for (int j = 0; j < 32; j += 8)
    out[(size_t)(bx + ty + j) * K + by + tx] = tile[tx][ty + j];
}

// ---------------------------------------------------------------------------
// Fragment loaders (A: row-major [row][k], A-operand K contiguous per lane half;
// B: Bt row-major [col][k], B-operand K contiguous per lane half)
// ---------------------------------------------------------------------------
__device__ __forceinline__ Frag load_afrag(const _Float16* p) {
  Frag f;
  const uint4* q = (const uint4*)p;
  f.q[0] = q[0];
  f.q[1] = q[1];  // +8 elements
  return f;
}
__device__ __forceinline__ Frag load_bfrag(const _Float16* p) {
  Frag f;
  const uint4* q = (const uint4*)p;
  f.q[0] = q[0];
  f.q[1] = q[1];  // +8 elements
  return f;
}

// ---------------------------------------------------------------------------
// GEMM core: C[32x64] per wave, software-pipelined (double-buffered frags).
// ---------------------------------------------------------------------------
__device__ __forceinline__ void gemm_core_32x64(
    const _Float16* __restrict__ A, const _Float16* __restrict__ Bt, int K,
    int m0, int n0, int ln, int hi, v8f acc[2][4]) {
  const _Float16* arow0 = A + (size_t)(m0 + ln) * K + hi * 16;
  const _Float16* arow1 = A + (size_t)(m0 + 16 + ln) * K + hi * 16;
  const _Float16* brow[4];
#pragma unroll
  for (int nt = 0; nt < 4; ++nt)
    brow[nt] = Bt + (size_t)(n0 + nt * 16 + ln) * K + hi * 16;

  Frag a[2], b[4];
  a[0] = load_afrag(arow0);
  a[1] = load_afrag(arow1);
#pragma unroll
  for (int nt = 0; nt < 4; ++nt) b[nt] = load_bfrag(brow[nt]);

#pragma unroll 2
  for (int k0 = 0; k0 < K; k0 += 32) {
    // issue next k-step's loads before consuming current fragments
    const int kn = (k0 + 32 < K) ? (k0 + 32) : 0;
    Frag an[2], bn[4];
    an[0] = load_afrag(arow0 + kn);
    an[1] = load_afrag(arow1 + kn);
#pragma unroll
    for (int nt = 0; nt < 4; ++nt) bn[nt] = load_bfrag(brow[nt] + kn);

#pragma unroll
    for (int nt = 0; nt < 4; ++nt)
#pragma unroll
      for (int mt = 0; mt < 2; ++mt)
        acc[mt][nt] = __builtin_amdgcn_wmma_f32_16x16x32_f16(
            false, a[mt].v, false, b[nt].v, (short)0, acc[mt][nt], false,
            false);

    a[0] = an[0];
    a[1] = an[1];
#pragma unroll
    for (int nt = 0; nt < 4; ++nt) b[nt] = bn[nt];
  }
}

// ---------------------------------------------------------------------------
// Projection GEMM: C[4096x1024] = A * B (Bt[n][k] f16). 8 waves/block.
// vmode 0: write [b][h][s][dh]   (Q, K)
// vmode 1: write [b][h][dh][s]   (V transposed for the PV A-operand)
// ---------------------------------------------------------------------------
__global__ __launch_bounds__(256) void gemm_qkv_kernel(
    const _Float16* __restrict__ A, const _Float16* __restrict__ Bt,
    _Float16* __restrict__ Out, int vmode) {
  const int lane = threadIdx.x & 31;
  const int wave = threadIdx.x >> 5;
  const int hi = lane >> 4;
  const int ln = lane & 15;
  const int m0 = (blockIdx.x * 8 + wave) * 32;
  const int n0 = blockIdx.y * 64;

  v8f acc[2][4] = {};
  gemm_core_32x64(A, Bt, 1024, m0, n0, ln, hi, acc);

#pragma unroll
  for (int mt = 0; mt < 2; ++mt)
#pragma unroll
    for (int nt = 0; nt < 4; ++nt) {
      const int n = n0 + nt * 16 + ln;
      const int hh = n >> 6, d = n & 63;
#pragma unroll
      for (int r = 0; r < 8; ++r) {
        const int m = m0 + mt * 16 + hi * 8 + r;
        const int bb = m >> 11, s = m & 2047;
        const _Float16 val = (_Float16)acc[mt][nt][r];
        const size_t idx =
            (vmode == 0) ? (((size_t)(bb * 16 + hh) * 2048 + s) * 64 + d)
                         : (((size_t)(bb * 16 + hh) * 64 + d) * 2048 + s);
        Out[idx] = val;
      }
    }
}

// ---------------------------------------------------------------------------
// Output GEMM: out[4096x1024] = ctx * Wo + bias, fp32 result.
// ---------------------------------------------------------------------------
__global__ __launch_bounds__(256) void gemm_out_kernel(
    const _Float16* __restrict__ A, const _Float16* __restrict__ Bt,
    const float* __restrict__ bias, float* __restrict__ Out) {
  const int lane = threadIdx.x & 31;
  const int wave = threadIdx.x >> 5;
  const int hi = lane >> 4;
  const int ln = lane & 15;
  const int m0 = (blockIdx.x * 8 + wave) * 32;
  const int n0 = blockIdx.y * 64;

  v8f acc[2][4] = {};
  gemm_core_32x64(A, Bt, 1024, m0, n0, ln, hi, acc);

#pragma unroll
  for (int mt = 0; mt < 2; ++mt)
#pragma unroll
    for (int nt = 0; nt < 4; ++nt) {
      const int n = n0 + nt * 16 + ln;
      const float bn = bias[n];
#pragma unroll
      for (int r = 0; r < 8; ++r) {
        const int m = m0 + mt * 16 + hi * 8 + r;
        Out[(size_t)m * 1024 + n] = acc[mt][nt][r] + bn;
      }
    }
}

// ---------------------------------------------------------------------------
// Causal flash attention. Grid: (S/128, B*H), 128 threads (4 waves).
// Wave handles 32 queries (2 tiles sharing K/V fragments); keys in 32-chunks.
// S^T = K @ Q^T so softmax stats are per-lane and the packed P^T C-layout is
// directly the B operand of ctx^T = V^T @ P^T (no LDS round-trips).
// All 16 chunk loads are issued up front so V loads overlap the score WMMAs
// and the softmax VALU section.
// ---------------------------------------------------------------------------
__global__ __launch_bounds__(128) void attn_kernel(
    const _Float16* __restrict__ Q, const _Float16* __restrict__ Kmat,
    const _Float16* __restrict__ Vt, _Float16* __restrict__ Ctx) {
  const int S = 2048, Dh = 64;
  const int lane = threadIdx.x & 31;
  const int wave = threadIdx.x >> 5;
  const int hi = lane >> 4;
  const int ln = lane & 15;
  const int bh = blockIdx.y;
  const int qbase = blockIdx.x * 128 + wave * 32;

  const _Float16* Qp = Q + (size_t)bh * S * Dh;
  const _Float16* Kp = Kmat + (size_t)bh * S * Dh;
  const _Float16* Vp = Vt + (size_t)bh * Dh * S;

  // Q as B-operand of S^T: lane = query, k-dim = dh (contiguous per half)
  Frag qf[2][2];
#pragma unroll
  for (int qt = 0; qt < 2; ++qt)
#pragma unroll
    for (int j = 0; j < 2; ++j)
      qf[qt][j] = load_bfrag(Qp + (size_t)(qbase + qt * 16 + ln) * Dh + j * 32 +
                             hi * 16);

  v8f ctx[2][4] = {};  // ctx^T per query tile: lane = query, regs = dh
  float m_i[2] = {-INFINITY, -INFINITY};
  float l_i[2] = {0.f, 0.f};
  const float fsc = 0.125f * 1.4426950408889634f;  // (1/sqrt(64)) * log2(e)

#pragma unroll 1
  for (int kc = 0; kc < qbase + 32; kc += 32) {
    if (kc + 32 < S)
      __builtin_prefetch(Kp + (size_t)(kc + 32 + ln) * Dh, 0, 1);

    // ---- all chunk loads up front: 4 K-frags + 4 V-frags (16 x b128) ----
    Frag ka[2][2];  // [key tile][dh frag], A-operand
#pragma unroll
    for (int t = 0; t < 2; ++t) {
      const _Float16* kb = Kp + (size_t)(kc + t * 16 + ln) * Dh + hi * 16;
      ka[t][0] = load_afrag(kb);
      ka[t][1] = load_afrag(kb + 32);
    }
    Frag vf[4];     // [dh tile], A-operand of ctx^T (k-dim = keys)
#pragma unroll
    for (int dt = 0; dt < 4; ++dt)
      vf[dt] = load_afrag(Vp + (size_t)(dt * 16 + ln) * S + kc + hi * 16);

    // ---- scores + online softmax per query tile ----
    Frag pf[2];  // P^T B-operand per query tile
#pragma unroll
    for (int qt = 0; qt < 2; ++qt) {
      const int qidx = qbase + qt * 16 + ln;
      float pe[2][8];
      float cmax = -INFINITY;
#pragma unroll
      for (int t = 0; t < 2; ++t) {
        v8f sc = {};
        sc = __builtin_amdgcn_wmma_f32_16x16x32_f16(
            false, ka[t][0].v, false, qf[qt][0].v, (short)0, sc, false, false);
        sc = __builtin_amdgcn_wmma_f32_16x16x32_f16(
            false, ka[t][1].v, false, qf[qt][1].v, (short)0, sc, false, false);
#pragma unroll
        for (int r = 0; r < 8; ++r) {
          const int key = kc + t * 16 + hi * 8 + r;
          const float sv = (key <= qidx) ? sc[r] : -INFINITY;
          pe[t][r] = sv;
          cmax = fmaxf(cmax, sv);
        }
      }
      // per-query reduction: other half of the 16 keys lives in lane^16
      cmax = fmaxf(cmax, __shfl_xor(cmax, 16, 32));
      const float m_new = fmaxf(m_i[qt], cmax);
      const float scale = exp2f((m_i[qt] - m_new) * fsc);  // 0 on first iter
      float rsum = 0.f;
#pragma unroll
      for (int t = 0; t < 2; ++t)
#pragma unroll
        for (int r = 0; r < 8; ++r) {
          const float pv = exp2f((pe[t][r] - m_new) * fsc);
          pe[t][r] = pv;
          rsum += pv;
        }
      rsum += __shfl_xor(rsum, 16, 32);
      l_i[qt] = l_i[qt] * scale + rsum;
      m_i[qt] = m_new;
#pragma unroll
      for (int dt = 0; dt < 4; ++dt)
#pragma unroll
        for (int r = 0; r < 8; ++r) ctx[qt][dt][r] *= scale;

      // pack pairs, exchange halves via shfl_xor(16) -> P^T B-operand
      unsigned int u0[4], u1[4];
#pragma unroll
      for (int j = 0; j < 4; ++j) {
        u0[j] = pack2h(pe[0][2 * j], pe[0][2 * j + 1]);
        u1[j] = pack2h(pe[1][2 * j], pe[1][2 * j + 1]);
      }
#pragma unroll
      for (int j = 0; j < 4; ++j) {
        const unsigned int x0 = (unsigned int)__shfl_xor((int)u0[j], 16, 32);
        const unsigned int x1 = (unsigned int)__shfl_xor((int)u1[j], 16, 32);
        pf[qt].u[j]     = hi ? x1    : u0[j];  // keys {0..7} | {16..23}
        pf[qt].u[4 + j] = hi ? u1[j] : x0;     // keys {8..15} | {24..31}
      }
    }

    // ---- ctx^T += V^T @ P^T ----
#pragma unroll
    for (int dt = 0; dt < 4; ++dt)
#pragma unroll
      for (int qt = 0; qt < 2; ++qt)
        ctx[qt][dt] = __builtin_amdgcn_wmma_f32_16x16x32_f16(
            false, vf[dt].v, false, pf[qt].v, (short)0, ctx[qt][dt], false,
            false);
  }

  // finalize: divide by l, store ctx row-major [b*S+s][h*64+dh] as f16
  const int b = bh >> 4, h = bh & 15;
#pragma unroll
  for (int qt = 0; qt < 2; ++qt) {
    const float inv_l = 1.0f / l_i[qt];
    _Float16* out =
        Ctx + ((size_t)(b * 2048 + qbase + qt * 16 + ln)) * 1024 + h * 64;
#pragma unroll
    for (int dt = 0; dt < 4; ++dt)
#pragma unroll
      for (int r = 0; r < 8; ++r)
        out[dt * 16 + hi * 8 + r] = (_Float16)(ctx[qt][dt][r] * inv_l);
  }
}

// ---------------------------------------------------------------------------
// Host launcher
// ---------------------------------------------------------------------------
extern "C" void kernel_launch(void* const* d_in, const int* in_sizes, int n_in,
                              void* d_out, int out_size, void* d_ws,
                              size_t ws_size, hipStream_t stream) {
  const float* x  = (const float*)d_in[0];
  const float* Wq = (const float*)d_in[1];
  const float* Wk = (const float*)d_in[2];
  const float* Wv = (const float*)d_in[3];
  const float* Wo = (const float*)d_in[4];
  const float* bo = (const float*)d_in[5];
  float* out = (float*)d_out;

  const size_t MB = 1u << 20;
  char* ws = (char*)d_ws;
  _Float16* xb  = (_Float16*)(ws + 0 * MB);    // 8 MB  [4096][1024]
  _Float16* WqT = (_Float16*)(ws + 8 * MB);    // 2 MB  [1024][1024]
  _Float16* WkT = (_Float16*)(ws + 10 * MB);
  _Float16* WvT = (_Float16*)(ws + 12 * MB);
  _Float16* WoT = (_Float16*)(ws + 14 * MB);
  _Float16* Qb  = (_Float16*)(ws + 16 * MB);   // 8 MB  [b][h][s][dh]
  _Float16* Kb  = (_Float16*)(ws + 24 * MB);   // 8 MB  [b][h][s][dh]
  _Float16* Vtb = (_Float16*)(ws + 32 * MB);   // 8 MB  [b][h][dh][s]
  _Float16* Ctx = (_Float16*)(ws + 40 * MB);   // 8 MB  [4096][1024]

  const int nx = 4096 * 1024;
  cvt_f16_kernel<<<nx / 256, 256, 0, stream>>>(x, xb, nx);
  dim3 tg(32, 32);  // 1024/32 x 1024/32
  transpose_f16_kernel<<<tg, 256, 0, stream>>>(Wq, WqT, 1024, 1024);
  transpose_f16_kernel<<<tg, 256, 0, stream>>>(Wk, WkT, 1024, 1024);
  transpose_f16_kernel<<<tg, 256, 0, stream>>>(Wv, WvT, 1024, 1024);
  transpose_f16_kernel<<<tg, 256, 0, stream>>>(Wo, WoT, 1024, 1024);

  dim3 gg(16, 16);  // 4096/(32*8) row groups, 1024/64 col groups
  gemm_qkv_kernel<<<gg, 256, 0, stream>>>(xb, WqT, Qb, 0);
  gemm_qkv_kernel<<<gg, 256, 0, stream>>>(xb, WkT, Kb, 0);
  gemm_qkv_kernel<<<gg, 256, 0, stream>>>(xb, WvT, Vtb, 1);

  attn_kernel<<<dim3(16, 32), 128, 0, stream>>>(Qb, Kb, Vtb, Ctx);

  gemm_out_kernel<<<gg, 256, 0, stream>>>(Ctx, WoT, bo, out);
}
